// KLD_317827580254
// MI455X (gfx1250) — matrix operand order repaired and checked by reading, not compiled, not run on hardware
//
#include <hip/hip_runtime.h>
#include <hip/hip_bf16.h>

typedef __attribute__((ext_vector_type(16))) _Float16 v16h;
typedef __attribute__((ext_vector_type(8)))  float    v8f;

#define NBIN 1000
#define Bsz  256
#define Hh   192
#define Ww   256
#define HW   (Hh*Ww)                 // 49152
#define TILES_PER_IMG ((Hh/16)*(Ww/16))  // 192
#define NTILES (Bsz*TILES_PER_IMG)       // 49152
#define WAVES_PER_BLOCK 8
#define NBLK1 (NTILES/WAVES_PER_BLOCK)   // 6144
#define PSTR 34   // f16 patch row stride (pad vs bank conflicts)

// normalized 11-tap gaussian, sigma=1.5 (matches reference _make_window 1-D factor)
__constant__ float GW[11] = {
    0.0010285f, 0.0075988f, 0.0360009f, 0.1093607f, 0.2130056f,
    0.2660117f,
    0.2130056f, 0.1093607f, 0.0360009f, 0.0075988f, 0.0010285f};

// ---------------------------------------------------------------------------
// Kernel 1: fused separable 11x11 conv (x5 quantities) via WMMA + SSIM partial.
// One wave = one 16x16 output tile.
//   Horizontal: t(32x16)      = Patch(32x32 as two 16x32 A tiles) x Band(32x16)
//   Vertical:   out^T(16x16)  = t^T(16x32, in-lane from the two D tiles) x Band(32x16)
// Band[k][n] = g[k-n] serves both passes; out^T is fine since we only need the
// elementwise SSIM sum. No LDS traffic between passes.
// ---------------------------------------------------------------------------
__global__ __launch_bounds__(256) void k_ssim(const float* __restrict__ img1,
                                              const float* __restrict__ img2,
                                              float* __restrict__ ssimPart) {
  __shared__ _Float16 sP1[WAVES_PER_BLOCK][32 * PSTR];
  __shared__ _Float16 sP2[WAVES_PER_BLOCK][32 * PSTR];

  const int lane = threadIdx.x & 31;
  const int wv   = threadIdx.x >> 5;
  const int gt   = blockIdx.x * WAVES_PER_BLOCK + wv;
  const int b    = gt / TILES_PER_IMG;
  const int rem  = gt % TILES_PER_IMG;
  const int h0   = (rem / (Ww / 16)) * 16;
  const int w0   = (rem % (Ww / 16)) * 16;
  const float* p1g = img1 + (size_t)b * HW;
  const float* p2g = img2 + (size_t)b * HW;

  // Stage 32x32 patch (valid 26x26 = tile + 5 halo each side), zero padded.
  // Iteration i loads row i, lanes give consecutive columns (coalesced).
  #pragma unroll 4
  for (int i = 0; i < 32; ++i) {
    const int r = i, c = lane;
    const int hh = h0 - 5 + r, ww = w0 - 5 + c;
    float a = 0.f, bb = 0.f;
    if (r < 26 && c < 26 && hh >= 0 && hh < Hh && ww >= 0 && ww < Ww) {
      const int idx = hh * Ww + ww;
      a  = p1g[idx];
      bb = p2g[idx];
    }
    sP1[wv][r * PSTR + c] = (_Float16)a;
    sP2[wv][r * PSTR + c] = (_Float16)bb;
  }

  const bool hi = lane >= 16;
  const int  nm = lane & 15;

  // Banded coefficient matrix, B layout (32x16): lane owns col n=l&15,
  // e<8 -> K=(hi?8:0)+e, e>=8 -> K=(hi?24:16)+(e-8); value g[K-n] inside the band.
  v16h bband;
  #pragma unroll
  for (int e = 0; e < 16; ++e) {
    const int kb = (e < 8) ? ((hi ? 8 : 0) + e) : ((hi ? 24 : 16) + (e - 8));
    const int db = kb - nm;
    bband[e] = (db >= 0 && db <= 10) ? (_Float16)GW[db] : (_Float16)0.f;
  }

  __syncthreads();  // patch visible to all lanes of the wave

  v8f acc[5];
  for (int q = 0; q < 5; ++q) {
    // ---- horizontal pass: two M row-blocks of the 26-row (padded 32) patch
    v8f c01[2];
    #pragma unroll
    for (int rb = 0; rb < 2; ++rb) {
      v16h a;
      #pragma unroll
      for (int e = 0; e < 16; ++e) {
        // A layout (16x32): lane owns row m=l&15; e<8 -> K=(hi?8:0)+e,
        // e>=8 -> K=(hi?24:16)+(e-8).
        const int ka  = (e < 8) ? ((hi ? 8 : 0) + e) : ((hi ? 24 : 16) + (e - 8));
        const int row = nm + rb * 16;
        const _Float16 x1 = sP1[wv][row * PSTR + ka];
        const _Float16 x2 = sP2[wv][row * PSTR + ka];
        _Float16 v;
        switch (q) {
          case 0:  v = x1;      break;
          case 1:  v = x2;      break;
          case 2:  v = x1 * x1; break;
          case 3:  v = x2 * x2; break;
          default: v = x1 * x2; break;
        }
        a[e] = v;
      }
      v8f c = {};
      c01[rb] = __builtin_amdgcn_wmma_f32_16x16x32_f16(false, a, false, bband,
                                                       (short)0, c, false, false);
    }

    // ---- vertical pass: A2 = t^T, built purely in-lane from the two D tiles.
    // For every lane: e<8 -> t[K][col]=c01[0][e], e>=8 -> c01[1][e-8].
    v16h a2;
    #pragma unroll
    for (int e = 0; e < 8; ++e) {
      a2[e]     = (_Float16)c01[0][e];
      a2[e + 8] = (_Float16)c01[1][e];
    }
    v8f z = {};
    acc[q] = __builtin_amdgcn_wmma_f32_16x16x32_f16(false, a2, false, bband,
                                                    (short)0, z, false, false);
  }

  // ---- SSIM map for this (transposed) tile + wave reduction
  const float C1c = 0.0001f, C2c = 0.0009f;
  float sum = 0.f;
  #pragma unroll
  for (int r = 0; r < 8; ++r) {
    const float mu1 = acc[0][r], mu2 = acc[1][r];
    const float x11 = acc[2][r], x22 = acc[3][r], x12 = acc[4][r];
    const float mu12 = mu1 * mu2;
    const float s1  = x11 - mu1 * mu1;
    const float s2  = x22 - mu2 * mu2;
    const float s12 = x12 - mu12;
    const float num = (2.f * mu12 + C1c) * (2.f * s12 + C2c);
    const float den = (mu1 * mu1 + mu2 * mu2 + C1c) * (s1 + s2 + C2c);
    sum += num / den;
  }
  #pragma unroll
  for (int off = 16; off > 0; off >>= 1) sum += __shfl_xor(sum, off, 32);
  if (lane == 0) ssimPart[gt] = sum;
}

// ---------------------------------------------------------------------------
// Kernel 2: per-(image,row) min/max + 1000-bin histogram in LDS.
// One block owns one row -> plain global stores, no global atomics.
// ---------------------------------------------------------------------------
__global__ __launch_bounds__(256) void k_hist(const float* __restrict__ img1,
                                              const float* __restrict__ img2,
                                              float* __restrict__ hist) {
  __shared__ float    sred[256];
  __shared__ unsigned sbin[NBIN];
  const int id  = blockIdx.x;                 // 0..511: [img1 rows | img2 rows]
  const int row = id & 255;
  const float* src = ((id < Bsz) ? img1 : img2) + (size_t)row * HW;
  const int t = threadIdx.x;

  float mn = 3.4e38f, mx = -3.4e38f;
  for (int i = t; i < HW; i += 256) {
    const float v = src[i];
    mn = fminf(mn, v);
    mx = fmaxf(mx, v);
  }
  sred[t] = mn; __syncthreads();
  for (int s = 128; s > 0; s >>= 1) { if (t < s) sred[t] = fminf(sred[t], sred[t + s]); __syncthreads(); }
  mn = sred[0]; __syncthreads();
  sred[t] = mx; __syncthreads();
  for (int s = 128; s > 0; s >>= 1) { if (t < s) sred[t] = fmaxf(sred[t], sred[t + s]); __syncthreads(); }
  mx = sred[0]; __syncthreads();

  for (int i = t; i < NBIN; i += 256) sbin[i] = 0u;
  __syncthreads();

  const float width = mx - mn;
  const float scale = (width > 0.f) ? (float)NBIN / width : 0.f;
  for (int i = t; i < HW; i += 256) {
    const float v = src[i];
    const float scaled = (width > 0.f) ? (v - mn) * scale : 0.f;
    int idx = (int)scaled;                     // trunc toward zero, like astype(int32)
    idx = idx < 0 ? 0 : (idx > NBIN - 1 ? NBIN - 1 : idx);
    atomicAdd(&sbin[idx], 1u);
  }
  __syncthreads();

  float* dst = hist + (size_t)id * NBIN;
  for (int i = t; i < NBIN; i += 256) dst[i] = (float)sbin[i];
}

// ---------------------------------------------------------------------------
// Kernel 3: per-row softmax of both hists + pointwise KL, block reduce.
// ---------------------------------------------------------------------------
__global__ __launch_bounds__(256) void k_kl(const float* __restrict__ hist,
                                            float* __restrict__ klRow) {
  __shared__ float sred[256];
  const int r = blockIdx.x, t = threadIdx.x;
  const float* h1 = hist + (size_t)r * NBIN;
  const float* h2 = hist + (size_t)(Bsz + r) * NBIN;

  float m1 = -3.4e38f, m2 = -3.4e38f;
  for (int i = t; i < NBIN; i += 256) { m1 = fmaxf(m1, h1[i]); m2 = fmaxf(m2, h2[i]); }
  sred[t] = m1; __syncthreads();
  for (int s = 128; s > 0; s >>= 1) { if (t < s) sred[t] = fmaxf(sred[t], sred[t + s]); __syncthreads(); }
  m1 = sred[0]; __syncthreads();
  sred[t] = m2; __syncthreads();
  for (int s = 128; s > 0; s >>= 1) { if (t < s) sred[t] = fmaxf(sred[t], sred[t + s]); __syncthreads(); }
  m2 = sred[0]; __syncthreads();

  float s1 = 0.f, s2 = 0.f;
  for (int i = t; i < NBIN; i += 256) { s1 += expf(h1[i] - m1); s2 += expf(h2[i] - m2); }
  sred[t] = s1; __syncthreads();
  for (int s = 128; s > 0; s >>= 1) { if (t < s) sred[t] += sred[t + s]; __syncthreads(); }
  s1 = sred[0]; __syncthreads();
  sred[t] = s2; __syncthreads();
  for (int s = 128; s > 0; s >>= 1) { if (t < s) sred[t] += sred[t + s]; __syncthreads(); }
  s2 = sred[0]; __syncthreads();

  const float inv1 = 1.f / s1, inv2 = 1.f / s2;
  float kl = 0.f;
  for (int i = t; i < NBIN; i += 256) {
    const float p1 = expf(h1[i] - m1) * inv1;
    const float p2 = expf(h2[i] - m2) * inv2;
    kl += expf(p2) * (p2 - p1);
  }
  sred[t] = kl; __syncthreads();
  for (int s = 128; s > 0; s >>= 1) { if (t < s) sred[t] += sred[t + s]; __syncthreads(); }
  if (t == 0) klRow[r] = sred[0];
}

// ---------------------------------------------------------------------------
// Kernel 4: final scalar combine.
// ---------------------------------------------------------------------------
__global__ __launch_bounds__(256) void k_final(const float* __restrict__ ssimPart,
                                               const float* __restrict__ klRow,
                                               float* __restrict__ out) {
  __shared__ float sred[256];
  const int t = threadIdx.x;
  float s = 0.f;
  for (int i = t; i < NTILES; i += 256) s += ssimPart[i];
  sred[t] = s; __syncthreads();
  for (int st = 128; st > 0; st >>= 1) { if (t < st) sred[t] += sred[t + st]; __syncthreads(); }
  const float ssimSum = sred[0]; __syncthreads();
  sred[t] = klRow[t]; __syncthreads();
  for (int st = 128; st > 0; st >>= 1) { if (t < st) sred[t] += sred[t + st]; __syncthreads(); }
  if (t == 0) {
    const float ssim = ssimSum / (float)((size_t)Bsz * HW);
    const float kl   = sred[0] / (float)Bsz;
    out[0] = (ssim > 0.75f) ? (kl + 1.f - ssim) : (1.f - ssim);
  }
}

extern "C" void kernel_launch(void* const* d_in, const int* in_sizes, int n_in,
                              void* d_out, int out_size, void* d_ws, size_t ws_size,
                              hipStream_t stream) {
  const float* img1 = (const float*)d_in[0];
  const float* img2 = (const float*)d_in[1];
  // d_in[2] (window) unused: coefficients are compile-time constants.
  float* ws       = (float*)d_ws;
  float* ssimPart = ws;                       // NTILES floats
  float* klRow    = ws + NTILES;              // 256 floats
  float* hist     = ws + NTILES + 256;        // 512*1000 floats (~2.2 MB total ws)

  k_ssim <<<NBLK1, 256, 0, stream>>>(img1, img2, ssimPart);
  k_hist <<<2 * Bsz, 256, 0, stream>>>(img1, img2, hist);
  k_kl   <<<Bsz, 256, 0, stream>>>(hist, klRow);
  k_final<<<1, 256, 0, stream>>>(ssimPart, klRow, (float*)d_out);
}
